// EsmFoldTriangularSelfAttentionBlock_82824149336557
// MI455X (gfx1250) — matrix operand order repaired and checked by reading, not compiled, not run on hardware
//
#include <hip/hip_runtime.h>
#include <hip/hip_bf16.h>
#include <math.h>

// ---------------------------------------------------------------------------
// EsmFold triangular self-attention block for gfx1250 (MI455X), wave32 WMMA.
// All matmuls route through v_wmma_f32_16x16x32_f16 (f16 inputs staged in LDS,
// fp32 accumulation). fp32 elementwise for LN / softmax / sigmoid gates.
// Staging: fixed-trip-count unconditional loops (no exec-mask guards),
// load-phase / store-phase split so multiple global_load_b128 stay in flight.
// ---------------------------------------------------------------------------

typedef __attribute__((ext_vector_type(16))) _Float16 v16h;
typedef __attribute__((ext_vector_type(8)))  float    v8f;
typedef __attribute__((ext_vector_type(4)))  _Float16 h4;

#define L_   256
#define CS_  1024
#define CZ_  128
#define HS_  32
#define HZ_  4

static __device__ __forceinline__ v8f wmma32(v16h a, v16h b, v8f c) {
  // D = A(16x32 f16) x B(32x16 f16) + C(16x16 f32)
  return __builtin_amdgcn_wmma_f32_16x16x32_f16(false, a, false, b, (short)0, c,
                                                false, false);
}

// A fragment (16x32, f16). Storage: row-major [m][k], leading dim ld (elems).
static __device__ __forceinline__ v16h load_a_frag(const _Float16* p, int ld) {
  int lane = threadIdx.x & 31;
  const _Float16* row = p + (lane & 15) * ld + ((lane >> 4) << 4);
  v16h a;
#pragma unroll
  for (int h = 0; h < 16; ++h) a[h] = row[h];
  return a;
}

// B fragment (32x16, f16). Storage: TRANSPOSED [n][k], leading dim ld.
static __device__ __forceinline__ v16h load_b_frag(const _Float16* p, int ld) {
  int lane = threadIdx.x & 31;
  const _Float16* row = p + (lane & 15) * ld + ((lane >> 4) << 4);
  v16h b;
#pragma unroll
  for (int h = 0; h < 16; ++h) b[h] = row[h];
  return b;
}

static __device__ __forceinline__ h4 cvt4(float4 f) {
  h4 hv;
  hv[0] = (_Float16)f.x; hv[1] = (_Float16)f.y;
  hv[2] = (_Float16)f.z; hv[3] = (_Float16)f.w;
  return hv;
}

// ---------------------------------------------------------------------------
// LayerNorm over last dim D. One block (256 threads) per row.
// ---------------------------------------------------------------------------
__global__ void __launch_bounds__(256)
k_ln(const float* __restrict__ x, const float* __restrict__ g,
     const float* __restrict__ b, float* __restrict__ y, int D) {
  __shared__ float red[256];
  size_t row = blockIdx.x;
  const float* xr = x + row * (size_t)D;
  float* yr = y + row * (size_t)D;
  int tid = threadIdx.x;
  float s = 0.f;
  for (int i = tid; i < D; i += 256) s += xr[i];
  red[tid] = s; __syncthreads();
  for (int o = 128; o > 0; o >>= 1) { if (tid < o) red[tid] += red[tid + o]; __syncthreads(); }
  float mean = red[0] / (float)D;
  __syncthreads();
  float v = 0.f;
  for (int i = tid; i < D; i += 256) { float d = xr[i] - mean; v += d * d; }
  red[tid] = v; __syncthreads();
  for (int o = 128; o > 0; o >>= 1) { if (tid < o) red[tid] += red[tid + o]; __syncthreads(); }
  float rstd = rsqrtf(red[0] / (float)D + 1e-5f);
  for (int i = tid; i < D; i += 256) yr[i] = (xr[i] - mean) * rstd * g[i] + b[i];
}

// ---------------------------------------------------------------------------
// Generic GEMM: C[M,N] = act(A[M,K] @ W[K,N] + bias) (+= if accum).
// 128 threads = 4 waves; 64x64 block tile; each wave a 32x32 tile (2x2 WMMA).
// Requires: M % 64 == 0, K % 32 == 0 (true for all uses). N arbitrary.
// act: 0 none, 1 relu, 2 sigmoid.
// ---------------------------------------------------------------------------
__global__ void __launch_bounds__(128)
k_gemm(const float* __restrict__ A, const float* __restrict__ W,
       const float* __restrict__ bias, float* __restrict__ C,
       int M, int N, int K, int act, int accum) {
  const int LD = 40;
  __shared__ _Float16 Ash[64 * LD];  // [m][k]
  __shared__ _Float16 Bsh[64 * LD];  // [n][k] (W transposed tile)
  int tid = threadIdx.x;             // blockDim.x == 128 always
  int wave = tid >> 5;
  int wm = (wave >> 1) << 5;
  int wn = (wave & 1) << 5;
  int row0 = blockIdx.y << 6;
  int col0 = blockIdx.x << 6;
  const bool wideN = (col0 + 64 <= N);  // block-uniform
  v8f acc[2][2] = {};
  for (int k0 = 0; k0 < K; k0 += 32) {
    // ---- A tile: 4 unconditional float4 loads in flight, then cvt+store
    {
      const float* Ab = A + (size_t)row0 * K + k0;
      float4 fa[4];
#pragma unroll
      for (int it = 0; it < 4; ++it) {
        int g = tid + it * 128;
        int m = g >> 3, kg = (g & 7) << 2;
        fa[it] = *(const float4*)(Ab + (size_t)m * K + kg);
      }
#pragma unroll
      for (int it = 0; it < 4; ++it) {
        int g = tid + it * 128;
        int m = g >> 3, kg = (g & 7) << 2;
        *(h4*)(Ash + m * LD + kg) = cvt4(fa[it]);
      }
    }
    // ---- W tile (transposed into [n][k])
    if (wideN) {
      float4 fw[4];
#pragma unroll
      for (int it = 0; it < 4; ++it) {
        int g = tid + it * 128;
        int k = g >> 4, n4 = (g & 15) << 2;
        fw[it] = *(const float4*)(W + (size_t)(k0 + k) * N + col0 + n4);
      }
#pragma unroll
      for (int it = 0; it < 4; ++it) {
        int g = tid + it * 128;
        int k = g >> 4, n4 = (g & 15) << 2;
        Bsh[(n4 + 0) * LD + k] = (_Float16)fw[it].x;
        Bsh[(n4 + 1) * LD + k] = (_Float16)fw[it].y;
        Bsh[(n4 + 2) * LD + k] = (_Float16)fw[it].z;
        Bsh[(n4 + 3) * LD + k] = (_Float16)fw[it].w;
      }
    } else {
      // branch-free clamp + select (small-N projections only)
#pragma unroll
      for (int gr = 0; gr < 2; ++gr) {
        float fv[8];
#pragma unroll
        for (int it = 0; it < 8; ++it) {
          int e = tid + (gr * 8 + it) * 128;
          int n = e >> 5, k = e & 31;
          int gn = col0 + n;
          int cl = gn < N ? gn : N - 1;
          fv[it] = W[(size_t)(k0 + k) * N + cl];
        }
#pragma unroll
        for (int it = 0; it < 8; ++it) {
          int e = tid + (gr * 8 + it) * 128;
          int n = e >> 5, k = e & 31;
          int gn = col0 + n;
          Bsh[n * LD + k] = (_Float16)(gn < N ? fv[it] : 0.f);
        }
      }
    }
    // Prefetch next K tile (global_prefetch_b8)
    if (k0 + 32 < K) {
      __builtin_prefetch(&A[(size_t)(row0 + (tid >> 1)) * K + k0 + 32], 0, 1);
      if (wideN)
        __builtin_prefetch(&W[(size_t)(k0 + 32) * N + col0 + (tid >> 1)], 0, 1);
    }
    __syncthreads();
    v16h a0 = load_a_frag(Ash + (wm + 0) * LD, LD);
    v16h a1 = load_a_frag(Ash + (wm + 16) * LD, LD);
    v16h b0 = load_b_frag(Bsh + (wn + 0) * LD, LD);
    v16h b1 = load_b_frag(Bsh + (wn + 16) * LD, LD);
    acc[0][0] = wmma32(a0, b0, acc[0][0]);
    acc[0][1] = wmma32(a0, b1, acc[0][1]);
    acc[1][0] = wmma32(a1, b0, acc[1][0]);
    acc[1][1] = wmma32(a1, b1, acc[1][1]);
    __syncthreads();
  }
  int lane = tid & 31;
  int cn = lane & 15, mo = (lane >> 4) << 3;
#pragma unroll
  for (int sm = 0; sm < 2; ++sm)
#pragma unroll
    for (int sn = 0; sn < 2; ++sn)
#pragma unroll
      for (int r = 0; r < 8; ++r) {
        int m = row0 + wm + sm * 16 + mo + r;
        int n = col0 + wn + sn * 16 + cn;
        if (n < N) {
          float v = acc[sm][sn][r];
          if (bias) v += bias[n];
          if (act == 1) v = fmaxf(v, 0.f);
          else if (act == 2) v = 1.f / (1.f + __expf(-v));
          size_t idx = (size_t)m * N + n;
          if (accum) C[idx] += v; else C[idx] = v;
        }
      }
}

// ---------------------------------------------------------------------------
// Triangle-mult einsum. outgoing: out[i,j,c] = sum_k a[i,k,c]*b[j,k,c]
//                       incoming: out[i,j,c] = sum_k a[k,i,c]*b[k,j,c]
// grid: (L/64, L/64, CZ); 128 threads. All dims exact -> no guards.
// ---------------------------------------------------------------------------
__global__ void __launch_bounds__(128)
k_tri_einsum(const float* __restrict__ a, const float* __restrict__ b,
             float* __restrict__ out, int outgoing) {
  const int LD = 40;
  __shared__ _Float16 Ash[64 * LD];
  __shared__ _Float16 Bsh[64 * LD];
  int tid = threadIdx.x;
  int wave = tid >> 5;
  int wm = (wave >> 1) << 5, wn = (wave & 1) << 5;
  int row0 = blockIdx.y << 6, col0 = blockIdx.x << 6;
  int c = blockIdx.z;
  v8f acc[2][2] = {};
  for (int k0 = 0; k0 < L_; k0 += 32) {
    // 16 elements per thread; grouped 8 loads in flight, then stores
#pragma unroll
    for (int gr = 0; gr < 2; ++gr) {
      float fa[8], fb[8];
#pragma unroll
      for (int it = 0; it < 8; ++it) {
        int e = tid + (gr * 8 + it) * 128;
        int m = e >> 5, k = e & 31;
        int kk = k0 + k;
        size_t ai = outgoing ? ((size_t)(row0 + m) * L_ + kk)
                             : ((size_t)kk * L_ + (row0 + m));
        size_t bi = outgoing ? ((size_t)(col0 + m) * L_ + kk)
                             : ((size_t)kk * L_ + (col0 + m));
        fa[it] = a[ai * CZ_ + c];
        fb[it] = b[bi * CZ_ + c];
      }
#pragma unroll
      for (int it = 0; it < 8; ++it) {
        int e = tid + (gr * 8 + it) * 128;
        int m = e >> 5, k = e & 31;
        Ash[m * LD + k] = (_Float16)fa[it];
        Bsh[m * LD + k] = (_Float16)fb[it];
      }
    }
    __syncthreads();
    v16h a0 = load_a_frag(Ash + (wm + 0) * LD, LD);
    v16h a1 = load_a_frag(Ash + (wm + 16) * LD, LD);
    v16h b0 = load_b_frag(Bsh + (wn + 0) * LD, LD);
    v16h b1 = load_b_frag(Bsh + (wn + 16) * LD, LD);
    acc[0][0] = wmma32(a0, b0, acc[0][0]);
    acc[0][1] = wmma32(a0, b1, acc[0][1]);
    acc[1][0] = wmma32(a1, b0, acc[1][0]);
    acc[1][1] = wmma32(a1, b1, acc[1][1]);
    __syncthreads();
  }
  int lane = tid & 31;
  int cn = lane & 15, mo = (lane >> 4) << 3;
#pragma unroll
  for (int sm = 0; sm < 2; ++sm)
#pragma unroll
    for (int sn = 0; sn < 2; ++sn)
#pragma unroll
      for (int r = 0; r < 8; ++r) {
        int i = row0 + wm + sm * 16 + mo + r;
        int j = col0 + wn + sn * 16 + cn;
        out[((size_t)i * L_ + j) * CZ_ + c] = acc[sm][sn][r];
      }
}

// ---------------------------------------------------------------------------
// Attention (shared by sequence + triangular). One block per (head, q-tile of
// 32, batch-row). 128 threads = 4 waves. Q/K/V staged f16 in LDS; QK^T and PV
// via WMMA; softmax in LDS.
// ---------------------------------------------------------------------------
__global__ void __launch_bounds__(128)
k_mha(const float* __restrict__ qp, const float* __restrict__ kp,
      const float* __restrict__ vp, long long qkv_bs, int rs, int hs,
      const float* __restrict__ bias, int bq, int bk,
      const float* __restrict__ mf, int row_mask,
      float* __restrict__ out, long long out_bs, int ors, int ohs,
      float scale) {
  const int QLD = 40, KLD = 40, VLD = 264, SLD = 264;
  __shared__ _Float16 Qs[32 * QLD];    // [q][c]
  __shared__ _Float16 Ks[L_ * KLD];    // [j][c]  (B-frag layout for QK^T)
  __shared__ _Float16 Vs[32 * VLD];    // [c][j]  (B-frag layout for PV)
  __shared__ _Float16 Ss[32 * SLD];    // scores / probs
  __shared__ float part[128];
  __shared__ float rowv[32];
  int tid = threadIdx.x;               // blockDim.x == 128 always
  int h = blockIdx.x;
  int q0 = blockIdx.y << 5;
  int bi = blockIdx.z;
  size_t boff = (size_t)bi * (size_t)qkv_bs;

  // Q: 32x32 = 256 float4 groups, 2 per thread
  {
    float4 fq[2];
#pragma unroll
    for (int it = 0; it < 2; ++it) {
      int g = tid + it * 128;
      int q = g >> 3, c4 = (g & 7) << 2;
      fq[it] = *(const float4*)(qp + boff + (size_t)(q0 + q) * rs + h * hs + c4);
    }
#pragma unroll
    for (int it = 0; it < 2; ++it) {
      int g = tid + it * 128;
      int q = g >> 3, c4 = (g & 7) << 2;
      float4 f = fq[it];
      f.x *= scale; f.y *= scale; f.z *= scale; f.w *= scale;
      *(h4*)(Qs + q * QLD + c4) = cvt4(f);
    }
  }
  // K/V: 256x32 each = 2048 float4 groups, 16 per thread; grouped by 4
#pragma unroll
  for (int gr = 0; gr < 4; ++gr) {
    float4 fk[4], fv[4];
#pragma unroll
    for (int it = 0; it < 4; ++it) {
      int g = tid + (gr * 4 + it) * 128;
      int j = g >> 3, c4 = (g & 7) << 2;
      size_t base = boff + (size_t)j * rs + h * hs + c4;
      fk[it] = *(const float4*)(kp + base);
      fv[it] = *(const float4*)(vp + base);
    }
#pragma unroll
    for (int it = 0; it < 4; ++it) {
      int g = tid + (gr * 4 + it) * 128;
      int j = g >> 3, c4 = (g & 7) << 2;
      *(h4*)(Ks + j * KLD + c4) = cvt4(fk[it]);
      Vs[(c4 + 0) * VLD + j] = (_Float16)fv[it].x;
      Vs[(c4 + 1) * VLD + j] = (_Float16)fv[it].y;
      Vs[(c4 + 2) * VLD + j] = (_Float16)fv[it].z;
      Vs[(c4 + 3) * VLD + j] = (_Float16)fv[it].w;
    }
  }
  __syncthreads();

  int wave = tid >> 5;
  int qsub = (wave >> 1) << 4;  // 0 / 16
  int ssub = (wave & 1) << 4;   // 0 / 16
  int lane = tid & 31;
  int cn = lane & 15, mo = (lane >> 4) << 3;

  v16h aq = load_a_frag(Qs + qsub * QLD, QLD);
  for (int jt = 0; jt < 8; ++jt) {
    v16h bkf = load_b_frag(Ks + (jt * 32 + ssub) * KLD, KLD);
    v8f s = {};
    s = wmma32(aq, bkf, s);
#pragma unroll
    for (int r = 0; r < 8; ++r) {
      int qg = q0 + qsub + mo + r;
      int jg = jt * 32 + ssub + cn;
      float v = s[r];
      if (bias) v += bias[(size_t)qg * bq + (size_t)jg * bk + h];
      v += row_mask ? (-1e9f * (1.f - mf[bi] * mf[jg]))
                    : ((mf[jg] > 0.5f) ? 0.f : -1e9f);
      Ss[(qsub + mo + r) * SLD + jg] = (_Float16)v;
    }
  }
  __syncthreads();

  {  // softmax: 4 threads per row, 64 columns each
    int row = tid >> 2, t4 = tid & 3, jb = t4 * 64;
    float mx = -3.0e38f;
    for (int j = 0; j < 64; ++j) mx = fmaxf(mx, (float)Ss[row * SLD + jb + j]);
    part[tid] = mx; __syncthreads();
    if (t4 == 0)
      rowv[row] = fmaxf(fmaxf(part[tid], part[tid + 1]),
                        fmaxf(part[tid + 2], part[tid + 3]));
    __syncthreads();
    float rm = rowv[row];
    float sum = 0.f;
    for (int j = 0; j < 64; ++j) {
      float e = __expf((float)Ss[row * SLD + jb + j] - rm);
      Ss[row * SLD + jb + j] = (_Float16)e;
      sum += e;
    }
    __syncthreads();
    part[tid] = sum; __syncthreads();
    if (t4 == 0)
      rowv[row] = part[tid] + part[tid + 1] + part[tid + 2] + part[tid + 3];
    __syncthreads();
    float inv = 1.f / rowv[row];
    for (int j = 0; j < 64; ++j)
      Ss[row * SLD + jb + j] = (_Float16)((float)Ss[row * SLD + jb + j] * inv);
  }
  __syncthreads();

  v8f oacc = {};
  for (int kt = 0; kt < 8; ++kt) {
    v16h ap = load_a_frag(Ss + qsub * SLD + kt * 32, SLD);
    v16h bv = load_b_frag(Vs + ssub * VLD + kt * 32, VLD);
    oacc = wmma32(ap, bv, oacc);
  }
  size_t bo = (size_t)bi * (size_t)out_bs;
#pragma unroll
  for (int r = 0; r < 8; ++r) {
    int qg = q0 + qsub + mo + r;
    int cg = ssub + cn;
    out[bo + (size_t)qg * ors + h * ohs + cg] = oacc[r];
  }
}

// ---------------------------------------------------------------------------
// Elementwise helpers
// ---------------------------------------------------------------------------
__global__ void k_maskf(const unsigned char* __restrict__ m, float* __restrict__ mf, int n) {
  int i = blockIdx.x * blockDim.x + threadIdx.x;
  if (i < n) mf[i] = m[i] ? 1.f : 0.f;
}
__global__ void k_mul(float* __restrict__ d, const float* __restrict__ a,
                      const float* __restrict__ b, size_t n) {
  size_t i = (size_t)blockIdx.x * blockDim.x + threadIdx.x;
  if (i < n) d[i] = a[i] * b[i];
}
__global__ void k_madd(float* __restrict__ d, const float* __restrict__ a,
                       const float* __restrict__ b, size_t n) {
  size_t i = (size_t)blockIdx.x * blockDim.x + threadIdx.x;
  if (i < n) d[i] += a[i] * b[i];
}
// d = mf[i]*mf[j]*g*p over pair tensor (i,j,c)
__global__ void k_gatepair(float* __restrict__ d, const float* __restrict__ g,
                           const float* __restrict__ p, const float* __restrict__ mf,
                           size_t n) {
  size_t idx = (size_t)blockIdx.x * blockDim.x + threadIdx.x;
  if (idx < n) {
    size_t ij = idx >> 7;                 // / CZ_
    int j = (int)(ij & (L_ - 1));
    int i = (int)(ij >> 8);
    d[idx] = mf[i] * mf[j] * g[idx] * p[idx];
  }
}
// cat[i,j,c] = c<64 ? q2[j,c]*k2[i,c] : q2[j,c-64]-k2[i,c-64];  s=[q2|k2] per row
__global__ void k_outer(const float* __restrict__ s, float* __restrict__ cat, size_t n) {
  size_t idx = (size_t)blockIdx.x * blockDim.x + threadIdx.x;
  if (idx < n) {
    int c = (int)(idx & (CZ_ - 1));
    size_t ij = idx >> 7;
    int j = (int)(ij & (L_ - 1));
    int i = (int)(ij >> 8);
    int cc = (c < 64) ? c : (c - 64);
    float q = s[(size_t)j * CZ_ + cc];
    float k = s[(size_t)i * CZ_ + 64 + cc];
    cat[idx] = (c < 64) ? (q * k) : (q - k);
  }
}
__global__ void k_transpose(float* __restrict__ dst, const float* __restrict__ src, size_t n) {
  size_t idx = (size_t)blockIdx.x * blockDim.x + threadIdx.x;
  if (idx < n) {
    int c = (int)(idx & (CZ_ - 1));
    size_t ij = idx >> 7;
    int j = (int)(ij & (L_ - 1));
    int i = (int)(ij >> 8);
    dst[idx] = src[((size_t)j * L_ + i) * CZ_ + c];
  }
}
__global__ void k_addT(float* __restrict__ dst, const float* __restrict__ src, size_t n) {
  size_t idx = (size_t)blockIdx.x * blockDim.x + threadIdx.x;
  if (idx < n) {
    int c = (int)(idx & (CZ_ - 1));
    size_t ij = idx >> 7;
    int j = (int)(ij & (L_ - 1));
    int i = (int)(ij >> 8);
    dst[idx] += src[((size_t)j * L_ + i) * CZ_ + c];
  }
}

// ---------------------------------------------------------------------------
// Host orchestration.
// Input flattening assumed: setup_inputs() dict order, nested params dict
// traversed depth-first in insertion order (83 leaves total).
// ---------------------------------------------------------------------------
extern "C" void kernel_launch(void* const* d_in, const int* in_sizes, int n_in,
                              void* d_out, int out_size, void* d_ws, size_t ws_size,
                              hipStream_t stream) {
  (void)in_sizes; (void)n_in; (void)out_size; (void)ws_size;
  constexpr int L = L_, CS = CS_, CZ = CZ_;
  constexpr size_t SEQ = (size_t)L * CS;        // 262144
  constexpr size_t LL = (size_t)L * L;          // 65536
  constexpr size_t PAIRN = LL * CZ;             // 8388608
  const float rscale = 0.17677669529663687f;    // 1/sqrt(32)

  const float* seq_in = (const float*)d_in[0];
  const float* pair_in = (const float*)d_in[1];
  const unsigned char* mask_in = (const unsigned char*)d_in[2];
  auto P = [&](int i) { return (const float*)d_in[i]; };
  // top-level params
  const float *ln1_g = P(3), *ln1_b = P(4);
  const float *p2s_ln_g = P(5), *p2s_ln_b = P(6), *p2s_W = P(7);
  const float *sa_Wqkv = P(8), *sa_Wg = P(9), *sa_bg = P(10);
  const float *sa_Wo = P(11), *sa_bo = P(12);
  const float *ms_ln_g = P(13), *ms_ln_b = P(14);
  const float *ms_W1 = P(15), *ms_b1 = P(16), *ms_W2 = P(17), *ms_b2 = P(18);
  const float *s2p_ln_g = P(19), *s2p_ln_b = P(20);
  const float *s2p_Wp = P(21), *s2p_bp = P(22), *s2p_Wo = P(23), *s2p_bo = P(24);
  const int tmo0 = 25, tmi0 = 41, tas0 = 57, tae0 = 67;
  const float *mp_ln_g = P(77), *mp_ln_b = P(78);
  const float *mp_W1 = P(79), *mp_b1 = P(80), *mp_W2 = P(81), *mp_b2 = P(82);

  float* out_seq = (float*)d_out;
  float* out_pair = out_seq + SEQ;

  // workspace layout (floats)
  float* w = (float*)d_ws;
  float* mf = w;                       // 256 (pad 1024)
  float* ybuf = w + 1024;              // SEQ
  float* qkv = ybuf + SEQ;             // 3*SEQ
  float* gbuf = qkv + 3 * SEQ;         // SEQ
  float* obuf = gbuf + SEQ;            // SEQ
  float* tbuf = obuf + SEQ;            // SEQ
  float* hseq = tbuf + SEQ;            // 4*SEQ
  float* Z  = hseq + 4 * SEQ;          // PAIR
  float* GA = Z + PAIRN;               // PAIR
  float* PA = GA + PAIRN;              // PAIR
  float* GB = PA + PAIRN;              // PAIR
  float* PB = GB + PAIRN;              // PAIR
  float* X  = PB + PAIRN;              // PAIR
  float* H4 = X + PAIRN;               // 4*PAIR (bias / cat / zt / MLP hidden)
  float* biasT = H4 + PAIRN;           // tri-bias region inside H4

  dim3 gp((unsigned)((PAIRN + 255) / 256));

  auto gemm = [&](const float* A, const float* W_, const float* bias, float* C,
                  int M, int N, int K, int act, int accum) {
    dim3 g((N + 63) / 64, (M + 63) / 64);
    k_gemm<<<g, 128, 0, stream>>>(A, W_, bias, C, M, N, K, act, accum);
  };
  auto ln = [&](const float* x, const float* g_, const float* b_, float* y,
                int rows, int D) {
    k_ln<<<rows, 256, 0, stream>>>(x, g_, b_, y, D);
  };

  // --- init outputs & mask ---
  hipMemcpyAsync(out_seq, seq_in, SEQ * sizeof(float), hipMemcpyDeviceToDevice, stream);
  hipMemcpyAsync(out_pair, pair_in, PAIRN * sizeof(float), hipMemcpyDeviceToDevice, stream);
  k_maskf<<<1, 256, 0, stream>>>(mask_in, mf, L);

  // --- pair -> sequence bias: biasS = ln(pair) @ p2s_W   (65536 x 32) in H4
  ln(pair_in, p2s_ln_g, p2s_ln_b, Z, (int)LL, CZ);
  gemm(Z, p2s_W, nullptr, H4, (int)LL, HS_, CZ, 0, 0);

  // --- sequence self-attention ---
  ln(out_seq, ln1_g, ln1_b, ybuf, L, CS);
  gemm(ybuf, sa_Wqkv, nullptr, qkv, L, 3 * CS, CS, 0, 0);
  {
    dim3 ga(HS_, L / 32, 1);
    k_mha<<<ga, 128, 0, stream>>>(qkv + 0, qkv + 32, qkv + 64,
                                  0LL, 3 * CS, 3 * 32,
                                  H4, L * HS_, HS_, mf, 0,
                                  obuf, 0LL, CS, 32, rscale);
  }
  gemm(ybuf, sa_Wg, sa_bg, gbuf, L, CS, CS, 2, 0);
  k_mul<<<(unsigned)((SEQ + 255) / 256), 256, 0, stream>>>(obuf, gbuf, obuf, SEQ);
  gemm(obuf, sa_Wo, sa_bo, out_seq, L, CS, CS, 0, 1);

  // --- sequence MLP ---
  ln(out_seq, ms_ln_g, ms_ln_b, ybuf, L, CS);
  gemm(ybuf, ms_W1, ms_b1, hseq, L, 4 * CS, CS, 1, 0);
  gemm(hseq, ms_W2, ms_b2, out_seq, L, CS, 4 * CS, 0, 1);

  // --- sequence -> pair ---
  ln(out_seq, s2p_ln_g, s2p_ln_b, ybuf, L, CS);
  gemm(ybuf, s2p_Wp, s2p_bp, tbuf, L, CZ, CS, 0, 0);   // s = [q2|k2] (256x128)
  k_outer<<<gp, 256, 0, stream>>>(tbuf, H4, PAIRN);    // cat(prod,diff)
  gemm(H4, s2p_Wo, s2p_bo, out_pair, (int)LL, CZ, CZ, 0, 1);

  // --- triangle multiplications ---
  auto tri_mul = [&](int b0, int outgoing) {
    ln(out_pair, P(b0 + 0), P(b0 + 1), Z, (int)LL, CZ);         // zl
    gemm(Z, P(b0 + 2), P(b0 + 3), GA, (int)LL, CZ, CZ, 2, 0);   // sig gate a
    gemm(Z, P(b0 + 4), P(b0 + 5), PA, (int)LL, CZ, CZ, 0, 0);   // proj a
    k_gatepair<<<gp, 256, 0, stream>>>(GA, GA, PA, mf, PAIRN);  // a
    gemm(Z, P(b0 + 6), P(b0 + 7), PB, (int)LL, CZ, CZ, 2, 0);   // sig gate b
    gemm(Z, P(b0 + 8), P(b0 + 9), GB, (int)LL, CZ, CZ, 0, 0);   // proj b
    k_gatepair<<<gp, 256, 0, stream>>>(PA, PB, GB, mf, PAIRN);  // b
    dim3 ge(L / 64, L / 64, CZ);
    k_tri_einsum<<<ge, 128, 0, stream>>>(GA, PA, X, outgoing);
    ln(X, P(b0 + 10), P(b0 + 11), GB, (int)LL, CZ);
    gemm(GB, P(b0 + 12), P(b0 + 13), X, (int)LL, CZ, CZ, 0, 0); // @Wz+bz
    gemm(Z, P(b0 + 14), P(b0 + 15), PB, (int)LL, CZ, CZ, 2, 0); // out gate
    k_madd<<<gp, 256, 0, stream>>>(out_pair, X, PB, PAIRN);
  };
  tri_mul(tmo0, 1);
  tri_mul(tmi0, 0);

  // --- triangular attentions ---
  auto tri_att = [&](int b0, const float* zsrc, float* dst, int accum) {
    ln(zsrc, P(b0 + 0), P(b0 + 1), Z, (int)LL, CZ);
    gemm(Z, P(b0 + 3), nullptr, GA, (int)LL, HZ_ * 32, CZ, 0, 0);  // q
    gemm(Z, P(b0 + 4), nullptr, PA, (int)LL, HZ_ * 32, CZ, 0, 0);  // k
    gemm(Z, P(b0 + 5), nullptr, GB, (int)LL, HZ_ * 32, CZ, 0, 0);  // v
    gemm(Z, P(b0 + 2), nullptr, biasT, (int)LL, HZ_, CZ, 0, 0);    // tri bias
    dim3 ga(HZ_, L / 32, L);
    k_mha<<<ga, 128, 0, stream>>>(GA, PA, GB,
                                  (long long)(L * HZ_ * 32), HZ_ * 32, 32,
                                  biasT, L * HZ_, HZ_, mf, 1,
                                  X, (long long)(L * HZ_ * 32), HZ_ * 32, 32,
                                  rscale);
    gemm(Z, P(b0 + 6), P(b0 + 7), PB, (int)LL, HZ_ * 32, CZ, 2, 0); // gate
    k_mul<<<gp, 256, 0, stream>>>(X, X, PB, PAIRN);
    gemm(X, P(b0 + 8), P(b0 + 9), dst, (int)LL, CZ, HZ_ * 32, 0, accum);
  };
  tri_att(tas0, out_pair, out_pair, 1);                // starting node
  k_transpose<<<gp, 256, 0, stream>>>(H4, out_pair, PAIRN);   // zt in H4
  tri_att(tae0, H4, Z, 0);                             // ending node (on zt)
  k_addT<<<gp, 256, 0, stream>>>(out_pair, Z, PAIRN);  // += delta^T

  // --- pair MLP ---
  ln(out_pair, mp_ln_g, mp_ln_b, Z, (int)LL, CZ);
  gemm(Z, mp_W1, mp_b1, H4, (int)LL, 4 * CZ, CZ, 1, 0);
  gemm(H4, mp_W2, mp_b2, out_pair, (int)LL, CZ, 4 * CZ, 0, 1);
}
